// SwinDecoder_59167469470046
// MI455X (gfx1250) — matrix operand order repaired and checked by reading, not compiled, not run on hardware
//
#include <hip/hip_runtime.h>
#include <hip/hip_bf16.h>

// ---------------------------------------------------------------------------
// CDNA5 (gfx1250) Swin-UNet decoder. wave32, WMMA bf16 16x16x32, f32 accum.
// ds_load_b128 fragment assembly, float4 staging with pointer-increment
// addressing (no in-loop 64-bit muls), packed v_cvt_pk_bf16_f32 conversion,
// pipelined GEMM mainloop, paired-row b32 transposed LDS commits.
// Requires M % 128 == 0 (true for every GEMM in this network).
// ---------------------------------------------------------------------------

typedef __attribute__((ext_vector_type(16))) __bf16 v16bf;
typedef __attribute__((ext_vector_type(8)))  __bf16 v8bf;
typedef __attribute__((ext_vector_type(4)))  __bf16 v4bf;
typedef __attribute__((ext_vector_type(2)))  __bf16 v2bf;
typedef __attribute__((ext_vector_type(8)))  float  v8f;

#if __has_builtin(__builtin_amdgcn_cvt_pk_bf16_f32)
__device__ __forceinline__ v2bf f2bf2(float a, float b) {
    return __builtin_amdgcn_cvt_pk_bf16_f32(a, b);
}
#else
__device__ __forceinline__ v2bf f2bf2(float a, float b) {
    v2bf p; p[0] = (__bf16)a; p[1] = (__bf16)b; return p;
}
#endif

__device__ __forceinline__ v4bf f2bf4(float4 v) {
    v2bf lo = f2bf2(v.x, v.y);
    v2bf hi = f2bf2(v.z, v.w);
    return __builtin_shufflevector(lo, hi, 0, 1, 2, 3);
}

// Two contiguous 16-byte LDS loads -> one 16-element bf16 fragment.
__device__ __forceinline__ v16bf load_frag2(const __bf16* p0, const __bf16* p1) {
    v8bf lo = *(const v8bf*)p0;
    v8bf hi = *(const v8bf*)p1;
    return __builtin_shufflevector(lo, hi, 0,1,2,3,4,5,6,7,8,9,10,11,12,13,14,15);
}

__device__ __forceinline__ float gelu_exact(float v) {
    return 0.5f * v * (1.0f + erff(v * 0.70710678118654752f));
}

// ---------------------------------------------------------------------------
// Generic GEMM:  OUT[M,N] = epilogue( A[M,K] @ W[K,N] )
// Block: 256 threads (8 waves). Tile BM=128 x BN=64, BK=32.
// ---------------------------------------------------------------------------
#define BM 128
#define BN 64
#define BK 32
#define APITCH 40   // bf16 elems; 80B row pitch (multiple of 16B)
#define BPITCH 40

__global__ __launch_bounds__(256) void gemm_wmma_kernel(
    const float* __restrict__ A, const float* __restrict__ Wt,
    const float* __restrict__ bias, const float* __restrict__ res,
    float* __restrict__ OUT,
    int M, int N, int K, int act, int accum,
    int pf, int co, int Ph, int Pw)
{
    __shared__ __bf16 As[BM][APITCH];
    __shared__ __bf16 BsT[BN][BPITCH];

    const int tid  = threadIdx.x;
    const int wave = tid >> 5;
    const int lane = tid & 31;
    const int hf   = lane >> 4;
    const int l16  = lane & 15;
    const int bm0  = blockIdx.y * BM;
    const int bn0  = blockIdx.x * BN;

    v8f acc[4];
    #pragma unroll
    for (int nt = 0; nt < 4; ++nt) { v8f z = {}; acc[nt] = z; }

    // Per-thread staging geometry (fixed across K loop):
    //  A: rows it*32 + (tid>>3), col group (tid&7)*4
    //  W: K-row pair {2kp,2kp+1}, col group n4*4
    const int ar  = tid >> 3;          // 0..31
    const int ac4 = tid & 7;           // 0..7
    const int kp  = tid >> 4;          // 0..15
    const int n4  = tid & 15;          // 0..15
    const int gn  = bn0 + n4 * 4;      // N % 16 == 0 -> float4 fully in/out
    const bool wok = (gn < N);

    const long aStep = 32L * K;
    const float* aCur = A + ((long)bm0 + ar) * K + ac4 * 4;
    const float* wCur = Wt + (long)(2 * kp) * N + (wok ? gn : 0);

    float4 ra[4];
    float4 rw0, rw1;
    const float4 f4z = make_float4(0.f, 0.f, 0.f, 0.f);

    auto load_slabs = [&]() {
        #pragma unroll
        for (int it = 0; it < 4; ++it)
            ra[it] = *(const float4*)(aCur + it * aStep);
        float4 t0 = *(const float4*)(wCur);
        float4 t1 = *(const float4*)(wCur + N);
        rw0 = wok ? t0 : f4z;
        rw1 = wok ? t1 : f4z;
        aCur += BK;
        wCur += (long)BK * N;
    };

    load_slabs();

    for (int k0 = 0; k0 < K; k0 += BK) {
        // Commit staged registers to LDS (bf16)
        #pragma unroll
        for (int it = 0; it < 4; ++it)
            *(v4bf*)&As[it * 32 + ar][ac4 * 4] = f2bf4(ra[it]);
        // W transposed commit: one b32 per column (pair of K rows packed)
        *(v2bf*)&BsT[n4 * 4 + 0][2 * kp] = f2bf2(rw0.x, rw1.x);
        *(v2bf*)&BsT[n4 * 4 + 1][2 * kp] = f2bf2(rw0.y, rw1.y);
        *(v2bf*)&BsT[n4 * 4 + 2][2 * kp] = f2bf2(rw0.z, rw1.z);
        *(v2bf*)&BsT[n4 * 4 + 3][2 * kp] = f2bf2(rw0.w, rw1.w);
        __syncthreads();

        // Issue next slab's global loads; they complete during the WMMAs.
        if (k0 + BK < K) load_slabs();
        // Prefetch the slab after that (global_prefetch_b8)
        if (k0 + 2 * BK < K) {
            __builtin_prefetch(aCur, 0, 1);
            __builtin_prefetch(wCur, 0, 1);
        }

        const __bf16* arow = &As[wave * 16 + l16][0];
        v16bf a = load_frag2(arow + 16 * hf, arow + 16 * hf + 8);
        #pragma unroll
        for (int nt = 0; nt < 4; ++nt) {
            const __bf16* brow = &BsT[nt * 16 + l16][0];
            v16bf b = load_frag2(brow + 16 * hf, brow + 16 * hf + 8);
            acc[nt] = __builtin_amdgcn_wmma_f32_16x16x32_bf16(
                false, a, false, b, (short)0, acc[nt], false, false);
        }
        __syncthreads();
    }

    // Epilogue (C/D layout: VGPR r -> row r + 8*hf, col = lane&15)
    #pragma unroll
    for (int nt = 0; nt < 4; ++nt) {
        #pragma unroll
        for (int r = 0; r < 8; ++r) {
            int m = bm0 + wave * 16 + r + 8 * hf;       // always < M (M%128==0)
            int n = bn0 + nt * 16 + l16;
            if (n < N) {
                float v = acc[nt][r];
                if (bias) v += bias[n];
                if (act)  v = gelu_exact(v);
                if (res)  v += res[(long)m * N + n];
                if (pf > 0) {
                    int fi  = n / (pf * co);
                    int rem = n - fi * pf * co;
                    int fj  = rem / co;
                    int c   = rem - fj * co;
                    int w   = m % Pw;
                    int hh  = (m / Pw) % Ph;
                    int b   = m / (Pw * Ph);
                    long off = ((((long)b * Ph + hh) * pf + fi) * ((long)Pw * pf)
                                + ((long)w * pf + fj)) * co + c;
                    if (accum) v += OUT[off];
                    OUT[off] = v;
                } else {
                    long off = (long)m * N + n;
                    if (accum) v += OUT[off];
                    OUT[off] = v;
                }
            }
        }
    }
}

// ---------------------------------------------------------------------------
// Windowed attention, one block (128 thr, 4 waves) per (window, head).
// N=64 tokens, hd=32.  q,k row-major; V stored d-major (transposed, committed
// as token-pair packed b32 stores) so both WMMA phases use contiguous b128
// fragment loads.
// ---------------------------------------------------------------------------
__device__ __forceinline__ int sw_grp(int u, int L) {
    return (u < L - 8) ? 0 : ((u < L - 4) ? 1 : 2);
}

__global__ __launch_bounds__(128) void attn_wmma_kernel(
    const float* __restrict__ QKV, const float* __restrict__ rpb,
    float* __restrict__ AO,
    int C, int heads, int nHw, int nWw, int shift, int Hdim, int Wdim)
{
    const int h   = blockIdx.x;
    const int win = blockIdx.y;
    const int nWW  = nHw * nWw;
    const int widx = win % nWW;
    const int wi = widx / nWw, wj = widx % nWw;

    __shared__ __bf16 qs[64][40];     // [token][d]
    __shared__ __bf16 ks[64][40];     // [token][d]  (acts as B^T for QK^T)
    __shared__ __bf16 vsT[32][72];    // [d][token]  (acts as B^T for P@V)
    __shared__ float  sc[64][68];     // scores; pitch 68 -> 16B-aligned rows
    __shared__ __bf16 ps[64][72];     // probabilities

    const int tid = threadIdx.x;
    const int wave = tid >> 5, lane = tid & 31;
    const int hf = lane >> 4, l16 = lane & 15;
    const float scale = 0.17677669529663687f;   // 32^-0.5
    const long base = (long)win * 64;
    const long C3 = 3L * C;

    // Stage q (pre-scaled) and k: 64 tokens x 32 d; pointer-increment loop.
    {
        const int r0 = tid >> 3, d4 = tid & 7;      // r = it*16 + r0
        const float* qP = QKV + (base + r0) * C3 + (long)h * 32 + d4 * 4;
        #pragma unroll
        for (int it = 0; it < 4; ++it) {
            float4 q4 = *(const float4*)(qP);
            float4 k4 = *(const float4*)(qP + C);
            q4.x *= scale; q4.y *= scale; q4.z *= scale; q4.w *= scale;
            *(v4bf*)&qs[it * 16 + r0][d4 * 4] = f2bf4(q4);
            *(v4bf*)&ks[it * 16 + r0][d4 * 4] = f2bf4(k4);
            qP += 16 * C3;
        }
    }
    // Stage V transposed: token-pair per thread, packed b32 commits.
    {
        const int tp0 = tid >> 3, d4 = tid & 7;     // tp = it*16 + tp0
        const float* vP = QKV + (base + 2 * tp0) * C3 + (long)h * 32 + d4 * 4 + 2 * C;
        #pragma unroll
        for (int it = 0; it < 2; ++it) {
            int tp = it * 16 + tp0;
            float4 v0 = *(const float4*)(vP);
            float4 v1 = *(const float4*)(vP + C3);
            *(v2bf*)&vsT[d4 * 4 + 0][2 * tp] = f2bf2(v0.x, v1.x);
            *(v2bf*)&vsT[d4 * 4 + 1][2 * tp] = f2bf2(v0.y, v1.y);
            *(v2bf*)&vsT[d4 * 4 + 2][2 * tp] = f2bf2(v0.z, v1.z);
            *(v2bf*)&vsT[d4 * 4 + 3][2 * tp] = f2bf2(v0.w, v1.w);
            vP += 32 * C3;
        }
    }
    __syncthreads();

    // Phase 1: scores = (q*scale) @ k^T  + rpb bias (+ shift mask)
    {
        const __bf16* arow = &qs[wave * 16 + l16][0];
        v16bf a = load_frag2(arow + 16 * hf, arow + 16 * hf + 8);
        #pragma unroll
        for (int nt = 0; nt < 4; ++nt) {
            const __bf16* brow = &ks[nt * 16 + l16][0];
            v16bf b = load_frag2(brow + 16 * hf, brow + 16 * hf + 8);
            v8f c = {};
            c = __builtin_amdgcn_wmma_f32_16x16x32_bf16(
                false, a, false, b, (short)0, c, false, false);
            #pragma unroll
            for (int r = 0; r < 8; ++r) {
                int m  = wave * 16 + r + 8 * hf;   // query position
                int mm = nt * 16 + l16;            // key position
                int i1 = m >> 3, j1 = m & 7, i2 = mm >> 3, j2 = mm & 7;
                int rel = (i1 - i2 + 7) * 15 + (j1 - j2 + 7);
                float v = c[r] + rpb[rel * heads + h];
                if (shift) {
                    int id1 = 3 * sw_grp(wi * 8 + i1, Hdim) + sw_grp(wj * 8 + j1, Wdim);
                    int id2 = 3 * sw_grp(wi * 8 + i2, Hdim) + sw_grp(wj * 8 + j2, Wdim);
                    if (id1 != id2) v -= 100.0f;
                }
                sc[m][mm] = v;
            }
        }
    }
    __syncthreads();

    // Softmax per row (vectorized row sweeps)
    if (tid < 64) {
        float mx = -1e30f;
        float buf[64];
        #pragma unroll
        for (int n = 0; n < 64; n += 4) {
            float4 v = *(const float4*)&sc[tid][n];
            buf[n] = v.x; buf[n+1] = v.y; buf[n+2] = v.z; buf[n+3] = v.w;
            mx = fmaxf(mx, fmaxf(fmaxf(v.x, v.y), fmaxf(v.z, v.w)));
        }
        float s = 0.0f;
        #pragma unroll
        for (int n = 0; n < 64; ++n) { buf[n] = __expf(buf[n] - mx); s += buf[n]; }
        float inv = 1.0f / s;
        #pragma unroll
        for (int n = 0; n < 64; n += 4) {
            float4 v = make_float4(buf[n] * inv, buf[n+1] * inv,
                                   buf[n+2] * inv, buf[n+3] * inv);
            *(v4bf*)&ps[tid][n] = f2bf4(v);
        }
    }
    __syncthreads();

    // Phase 2: out = P(64x64) @ V(64x32)
    #pragma unroll
    for (int nt = 0; nt < 2; ++nt) {
        v8f c = {};
        #pragma unroll
        for (int kc = 0; kc < 2; ++kc) {
            const __bf16* arow = &ps[wave * 16 + l16][0];
            v16bf a = load_frag2(arow + kc * 32 + 16 * hf,
                                 arow + kc * 32 + 16 * hf + 8);
            const __bf16* brow = &vsT[nt * 16 + l16][0];
            v16bf b = load_frag2(brow + kc * 32 + 16 * hf,
                                 brow + kc * 32 + 16 * hf + 8);
            c = __builtin_amdgcn_wmma_f32_16x16x32_bf16(
                false, a, false, b, (short)0, c, false, false);
        }
        #pragma unroll
        for (int r = 0; r < 8; ++r) {
            int m = wave * 16 + r + 8 * hf;
            int d = nt * 16 + l16;
            AO[(base + m) * (long)C + (long)h * 32 + d] = c[r];
        }
    }
}

// ---------------------------------------------------------------------------
// LayerNorm over last dim C (in-place safe, float4 sweeps). 128 thr per row.
// ---------------------------------------------------------------------------
__global__ __launch_bounds__(128) void ln_kernel(
    const float* __restrict__ x, float* __restrict__ out,
    const float* __restrict__ g, const float* __restrict__ b, int C)
{
    __shared__ float s1[128], s2[128];
    const long row = blockIdx.x;
    const float* xr = x + row * (long)C;
    float sum = 0.0f, sq = 0.0f;
    for (int c = threadIdx.x * 4; c < C; c += 512) {
        float4 v = *(const float4*)(xr + c);
        sum += v.x + v.y + v.z + v.w;
        sq  += v.x * v.x + v.y * v.y + v.z * v.z + v.w * v.w;
    }
    s1[threadIdx.x] = sum; s2[threadIdx.x] = sq;
    __syncthreads();
    for (int st = 64; st > 0; st >>= 1) {
        if (threadIdx.x < st) { s1[threadIdx.x] += s1[threadIdx.x + st]; s2[threadIdx.x] += s2[threadIdx.x + st]; }
        __syncthreads();
    }
    float mu  = s1[0] / C;
    float var = s2[0] / C - mu * mu;
    float inv = rsqrtf(var + 1e-5f);
    float* orow = out + row * (long)C;
    for (int c = threadIdx.x * 4; c < C; c += 512) {
        float4 v  = *(const float4*)(xr + c);
        float4 gg = *(const float4*)(g + c);
        float4 bb = *(const float4*)(b + c);
        float4 o;
        o.x = (v.x - mu) * inv * gg.x + bb.x;
        o.y = (v.y - mu) * inv * gg.y + bb.y;
        o.z = (v.z - mu) * inv * gg.z + bb.z;
        o.w = (v.w - mu) * inv * gg.w + bb.w;
        *(float4*)(orow + c) = o;
    }
}

// ---------------------------------------------------------------------------
// Window partition with cyclic shift (roll by -shift fused), float4/thread.
// ---------------------------------------------------------------------------
__global__ void win_part_kernel(const float* __restrict__ x, float* __restrict__ y,
                                int Bn, int H, int W, int C, int shift)
{
    int C4 = C >> 2;
    long total = (long)Bn * H * W * C4;
    long idx = (long)blockIdx.x * blockDim.x + threadIdx.x;
    if (idx >= total) return;
    int c4 = (int)(idx % C4); long t = idx / C4;
    int j = (int)(t & 7);
    int i = (int)((t >> 3) & 7);
    long wt = t >> 6;
    int nWw = W >> 3, nHw = H >> 3;
    int wj = (int)(wt % nWw); long r2 = wt / nWw;
    int wi = (int)(r2 % nHw);
    int b  = (int)(r2 / nHw);
    int sh = (wi * 8 + i + shift) % H;
    int sw = (wj * 8 + j + shift) % W;
    float4 v = *(const float4*)(x + (((long)b * H + sh) * W + sw) * C + c4 * 4);
    *(float4*)(y + t * (long)C + c4 * 4) = v;
}

// Window reverse + inverse roll + residual add:  x += reverse(p), float4/thread
__global__ void win_rev_res_kernel(float* __restrict__ x, const float* __restrict__ p,
                                   int Bn, int H, int W, int C, int shift)
{
    int C4 = C >> 2;
    long total = (long)Bn * H * W * C4;
    long idx = (long)blockIdx.x * blockDim.x + threadIdx.x;
    if (idx >= total) return;
    int c4 = (int)(idx % C4); long t = idx / C4;
    int w = (int)(t % W); long r2 = t / W;
    int hh = (int)(r2 % H);
    int b  = (int)(r2 / H);
    int hs  = (hh + H - shift) % H;
    int ws2 = (w + W - shift) % W;
    int wi = hs >> 3, i = hs & 7, wj = ws2 >> 3, j = ws2 & 7;
    int nWw = W >> 3, nHw = H >> 3;
    long tok = (((long)b * nHw + wi) * nWw + wj) * 64 + i * 8 + j;
    float4 a = *(float4*)(x + t * (long)C + c4 * 4);
    float4 v = *(const float4*)(p + tok * (long)C + c4 * 4);
    a.x += v.x; a.y += v.y; a.z += v.z; a.w += v.w;
    *(float4*)(x + t * (long)C + c4 * 4) = a;
}

// ---------------------------------------------------------------------------
// Host orchestration
// ---------------------------------------------------------------------------
extern "C" void kernel_launch(void* const* d_in, const int* in_sizes, int n_in,
                              void* d_out, int out_size, void* d_ws, size_t ws_size,
                              hipStream_t stream)
{
    (void)in_sizes; (void)n_in; (void)out_size; (void)ws_size;

    const float* x3  = (const float*)d_in[0];
    const float* x2s = (const float*)d_in[1];
    const float* x1s = (const float*)d_in[2];
    const float* x0s = (const float*)d_in[3];

    // params flattened as JAX pytree (dict keys sorted alphabetically)
    const int P = 4;
    auto prm = [&](int i) { return (const float*)d_in[P + i]; };

    struct Blk { const float *fc1_b,*fc1_w,*fc2_b,*fc2_w,*n1_b,*n1_g,*n2_b,*n2_g,
                             *proj_b,*proj_w,*qkv_b,*qkv_w,*rpb; };
    auto getBlk = [&](int s) {
        Blk k;
        k.fc1_b = prm(s+0);  k.fc1_w = prm(s+1);  k.fc2_b = prm(s+2);  k.fc2_w = prm(s+3);
        k.n1_b  = prm(s+4);  k.n1_g  = prm(s+5);  k.n2_b  = prm(s+6);  k.n2_g  = prm(s+7);
        k.proj_b= prm(s+8);  k.proj_w= prm(s+9);  k.qkv_b = prm(s+10); k.qkv_w = prm(s+11);
        k.rpb   = prm(s+12);
        return k;
    };
    Blk msa0_0 = getBlk(0),  msa0_1 = getBlk(13);
    Blk msa1_0 = getBlk(26), msa1_1 = getBlk(39);
    Blk msa2_0 = getBlk(52), msa2_1 = getBlk(65);

    struct PE { const float *b,*beta,*g,*w; };
    auto getPE = [&](int s) { PE p; p.b=prm(s); p.beta=prm(s+1); p.g=prm(s+2); p.w=prm(s+3); return p; };
    PE pe0 = getPE(78), pe1 = getPE(82), pe2 = getPE(86), pelast = getPE(90);

    struct SK { const float *b,*w; };
    SK sk0{prm(94), prm(95)}, sk1{prm(96), prm(97)}, sk2{prm(98), prm(99)};

    // Workspace layout (floats)
    float* wsf = (float*)d_ws;
    size_t off = 0;
    auto walloc = [&](size_t n) { float* p = wsf + off; off += n; return p; };
    float* X   = walloc(3200000);   // feature map, max 8*64*64*96
    float* Y   = walloc(3200000);   // LN / window / proj buffer
    float* QKV = walloc(9500000);   // max 32768*288
    float* AO  = walloc(3200000);   // attention output
    float* H1  = walloc(12600000);  // max 32768*384
    float* X2  = walloc(3200000);   // patch-expand output

    auto gemm = [&](const float* A, const float* Wt, const float* bias, const float* res,
                    float* OUT, int M, int N, int K, int act, int accum,
                    int pf, int co, int Ph, int Pw) {
        dim3 g((N + BN - 1) / BN, (M + BM - 1) / BM);
        gemm_wmma_kernel<<<g, 256, 0, stream>>>(A, Wt, bias, res, OUT,
                                                M, N, K, act, accum, pf, co, Ph, Pw);
    };
    auto ln = [&](const float* x, float* o, const float* g, const float* b, long rows, int C) {
        ln_kernel<<<(unsigned)rows, 128, 0, stream>>>(x, o, g, b, C);
    };

    auto swin = [&](float* Xb, const Blk& bp, int Bn, int H, int W, int C,
                    int heads, int shift) {
        long T = (long)Bn * H * W;
        long total4 = T * (C / 4);
        int nHw = H / 8, nWw = W / 8;
        unsigned gb = (unsigned)((total4 + 255) / 256);
        // partition (+roll) then LN in place (LN is per-token -> commutes)
        win_part_kernel<<<gb, 256, 0, stream>>>(Xb, Y, Bn, H, W, C, shift);
        ln(Y, Y, bp.n1_g, bp.n1_b, T, C);
        gemm(Y, bp.qkv_w, bp.qkv_b, nullptr, QKV, (int)T, 3 * C, C, 0, 0, 0, 0, 0, 0);
        attn_wmma_kernel<<<dim3(heads, (unsigned)(Bn * nHw * nWw)), 128, 0, stream>>>(
            QKV, bp.rpb, AO, C, heads, nHw, nWw, shift, H, W);
        gemm(AO, bp.proj_w, bp.proj_b, nullptr, Y, (int)T, C, C, 0, 0, 0, 0, 0, 0);
        win_rev_res_kernel<<<gb, 256, 0, stream>>>(Xb, Y, Bn, H, W, C, shift);
        // MLP with residual fused in fc2 epilogue
        ln(Xb, Y, bp.n2_g, bp.n2_b, T, C);
        gemm(Y, bp.fc1_w, bp.fc1_b, nullptr, H1, (int)T, 4 * C, C, 1, 0, 0, 0, 0, 0);
        gemm(H1, bp.fc2_w, bp.fc2_b, Xb, Xb, (int)T, C, 4 * C, 0, 0, 0, 0, 0, 0);
    };

    // ---- Stage 2: pe2 (768 -> 2x2x384), concat x_2s, skip 768->384 ----
    gemm(x3, pe2.w, pe2.b, nullptr, X2, 512, 1536, 768, 0, 0, 2, 384, 8, 8);
    ln(X2, X2, pe2.g, pe2.beta, 8L * 16 * 16, 384);
    gemm(X2,  sk2.w,             sk2.b,   nullptr, X, 2048, 384, 384, 0, 0, 0, 0, 0, 0);
    gemm(x2s, sk2.w + 384 * 384, nullptr, nullptr, X, 2048, 384, 384, 0, 1, 0, 0, 0, 0);
    swin(X, msa2_0, 8, 16, 16, 384, 12, 0);
    swin(X, msa2_1, 8, 16, 16, 384, 12, 4);

    // ---- Stage 1 ----
    gemm(X, pe1.w, pe1.b, nullptr, X2, 2048, 768, 384, 0, 0, 2, 192, 16, 16);
    ln(X2, X2, pe1.g, pe1.beta, 8L * 32 * 32, 192);
    gemm(X2,  sk1.w,             sk1.b,   nullptr, X, 8192, 192, 192, 0, 0, 0, 0, 0, 0);
    gemm(x1s, sk1.w + 192 * 192, nullptr, nullptr, X, 8192, 192, 192, 0, 1, 0, 0, 0, 0);
    swin(X, msa1_0, 8, 32, 32, 192, 6, 0);
    swin(X, msa1_1, 8, 32, 32, 192, 6, 4);

    // ---- Stage 0 ----
    gemm(X, pe0.w, pe0.b, nullptr, X2, 8192, 384, 192, 0, 0, 2, 96, 32, 32);
    ln(X2, X2, pe0.g, pe0.beta, 8L * 64 * 64, 96);
    gemm(X2,  sk0.w,           sk0.b,   nullptr, X, 32768, 96, 96, 0, 0, 0, 0, 0, 0);
    gemm(x0s, sk0.w + 96 * 96, nullptr, nullptr, X, 32768, 96, 96, 0, 1, 0, 0, 0, 0);
    swin(X, msa0_0, 8, 64, 64, 96, 3, 0);
    swin(X, msa0_1, 8, 64, 64, 96, 3, 4);

    // ---- Final 4x patch expand -> d_out, LN in place ----
    float* out = (float*)d_out;
    gemm(X, pelast.w, pelast.b, nullptr, out, 32768, 1536, 96, 0, 0, 4, 96, 64, 64);
    ln(out, out, pelast.g, pelast.beta, 8L * 256 * 256, 96);
}
